// Attention_24919400252055
// MI455X (gfx1250) — matrix operand order repaired and checked by reading, not compiled, not run on hardware
//
#include <hip/hip_runtime.h>
#include <hip/hip_bf16.h>

// ---------------------------------------------------------------------------
// Single-head attention with softmax over the QUERY axis, for MI455X (gfx1250).
//   q,k,v = x@W + b          (f16 WMMA, f32 accum)
//   s     = q.k^T / sqrt(96)
//   P     = softmax_over_q(s)   (per-key-column stats, two-pass recompute)
//   out   = P @ v
// Tensor-Data-Mover (TDM) double-buffered LDS staging for operands shared by
// all waves of a workgroup (W tiles in the QKV GEMM; K/V tiles in the PV
// kernel), drained with s_wait_tensorcnt. 6-arg clang-23 TDM builtin.
// ---------------------------------------------------------------------------

typedef __attribute__((ext_vector_type(16))) _Float16 v16h;
typedef __attribute__((ext_vector_type(8)))  _Float16 h8;
typedef __attribute__((ext_vector_type(8)))  float    v8f;
typedef __attribute__((ext_vector_type(4)))  unsigned int u32x4;
typedef __attribute__((ext_vector_type(8)))  int          i32x8;
typedef __attribute__((ext_vector_type(4)))  int          i32x4;

#define ADIM   768
#define HD     96
#define ABATCH 4
#define ASEQ   4096
#define NTOK   (ABATCH * ASEQ)          // 16384
#define QKSCALE 0.10206207261596575f    // 1/sqrt(96)

#if defined(__has_builtin)
#if __has_builtin(__builtin_amdgcn_tensor_load_to_lds) && \
    __has_builtin(__builtin_amdgcn_s_wait_tensorcnt)
#define ATTN_HAS_TDM 1
#endif
#endif

// ---- WMMA wrapper: D = A(16x32 f16) * B(32x16 f16) + C(16x16 f32) ----------
__device__ inline v8f wmma16(v16h a, v16h b, v8f c) {
    return __builtin_amdgcn_wmma_f32_16x16x32_f16(
        /*neg_a=*/false, a, /*neg_b=*/false, b,
        /*c_mod=*/(short)0, c, /*reuse_a=*/false, /*reuse_b=*/false);
}

// ---- A fragment loader: 16x32 f16 tile from row-major matrix ---------------
// ISA layout (16-bit A 16x32): lanes 0-15: row=lane, K {0..7,16..23};
//                              lanes 16-31: row=lane-16, K {8..15,24..31}.
__device__ inline v16h load_a(const _Float16* base, int ld,
                              int row0, int col0, int lane) {
    const int r  = row0 + (lane & 15);
    const int cb = col0 + ((lane >> 4) << 3);   // +0 or +8
    const _Float16* p = base + (size_t)r * ld + cb;
    union { v16h v; h8 h[2]; } u;
    u.h[0] = *(const h8*)(p);        // K cb .. cb+7
    u.h[1] = *(const h8*)(p + 16);   // K cb+16 .. cb+23
    return u.v;
}

// ---- B fragment loader: 32x16 f16 tile, element (k,n) at base[n*ldn + k] ---
// ISA layout (16-bit B 32x16): lanes 0-15: col=lane, K {0..7,16..23};
//                              lanes 16-31: col=lane-16, K {8..15,24..31}.
__device__ inline v16h load_b(const _Float16* base, int ldn,
                              int n0, int k0, int lane) {
    const int n  = n0 + (lane & 15);
    const int kb = k0 + ((lane >> 4) << 3);     // +0 or +8
    const _Float16* p = base + (size_t)n * ldn + kb;
    union { v16h v; h8 h[2]; } u;
    u.h[0] = *(const h8*)(p);        // K kb .. kb+7
    u.h[1] = *(const h8*)(p + 16);   // K kb+16 .. kb+23
    return u.v;
}

// ---- LDS byte offset of a __shared__ pointer -------------------------------
// Flat LDS-aperture addresses carry the LDS byte offset in the low 32 bits
// (ISA 10.2).
__device__ inline unsigned lds_off(const void* p) {
    return (unsigned)(uintptr_t)p;
}

#if ATTN_HAS_TDM
// ---- TDM: 2-D tile (f16) global -> LDS, D# per CDNA5 ISA -------
// tile_d0 elements contiguous per row, tile_d1 rows, row stride s0 elements.
// Tensor dims set tight (== tile dims) so nothing is OOB.
__device__ inline void tdm_load_2d_f16(unsigned lds_addr, const void* gptr,
                                       unsigned tile_d0, unsigned tile_d1,
                                       unsigned stride_d0) {
    unsigned long long ga = (unsigned long long)(uintptr_t)gptr;
    u32x4 g0;
    g0[0] = 1u;                                          // count=1 (valid, user)
    g0[1] = lds_addr;                                    // lds_addr (bytes)
    g0[2] = (unsigned)(ga & 0xffffffffu);                // global_addr[31:0]
    g0[3] = (unsigned)((ga >> 32) & 0x01ffffffu)         // global_addr[56:32]
          | (2u << 30);                                  // type = 2 ("image")
    i32x8 g1;
    g1[0] = (int)(1u << 16);                             // data_size=1 -> 2 bytes
    g1[1] = (int)((tile_d0 & 0xffffu) << 16);            // tensor_dim0[15:0]
    g1[2] = (int)(((tile_d0 >> 16) & 0xffffu)            // tensor_dim0[31:16]
          | ((tile_d1 & 0xffffu) << 16));                // tensor_dim1[15:0]
    g1[3] = (int)(((tile_d1 >> 16) & 0xffffu)            // tensor_dim1[31:16]
          | ((tile_d0 & 0xffffu) << 16));                // tile_dim0
    g1[4] = (int)(tile_d1 & 0xffffu);                    // tile_dim1 (tile_dim2=0)
    g1[5] = (int)stride_d0;                              // tensor_dim0_stride[31:0]
    g1[6] = 0;                                           // stride hi / dim1_stride lo
    g1[7] = 0;
    i32x4 z4 = {0, 0, 0, 0};
    i32x8 z8 = {0, 0, 0, 0, 0, 0, 0, 0};
    // clang-23 / therock-10.0 arity: (g0, g1, g2, g3, g4, cpol)
    __builtin_amdgcn_tensor_load_to_lds(g0, g1, z4, z4, z8, 0);
}
#endif

// ---- fallback cooperative copy: rows x row_halves halves, arbitrary src stride -----
__device__ inline void coop_copy_rows(_Float16* dst, const _Float16* src,
                                      int rows, int row_halves, int src_stride,
                                      int tid, int nthr) {
    const int chunks_per_row = row_halves >> 3;
    const int total = rows * chunks_per_row;
    for (int i = tid; i < total; i += nthr) {
        const int r = i / chunks_per_row;
        const int c = (i - r * chunks_per_row) << 3;
        *(h8*)(dst + (size_t)r * row_halves + c) =
            *(const h8*)(src + (size_t)r * src_stride + c);
    }
}

// ---------------------------------------------------------------------------
// Kernel 1: f32 -> f16 conversion of x, and f16 TRANSPOSE of Wq/Wk/Wv.
//   xh[i]            = (f16) x[i]
//   wt[m][n*768 + k] = (f16) W_m[k*96 + n]     (n-major for B-fragment loads)
// ---------------------------------------------------------------------------
__global__ void attn_convert_kernel(const float* __restrict__ x,
                                    const float* __restrict__ Wq,
                                    const float* __restrict__ Wk,
                                    const float* __restrict__ Wv,
                                    _Float16* __restrict__ xh,
                                    _Float16* __restrict__ wt) {
    const int NX = NTOK * ADIM;
    const int NW = ADIM * HD;
    const int total = NX + 3 * NW;
    for (int i = blockIdx.x * blockDim.x + threadIdx.x; i < total;
         i += gridDim.x * blockDim.x) {
        if (i < NX) {
            xh[i] = (_Float16)x[i];
        } else {
            int j = i - NX;
            int m = j / NW;
            int r = j - m * NW;
            int n = r / ADIM;          // output column (0..95)
            int k = r - n * ADIM;      // reduction index (0..767)
            const float* W = (m == 0) ? Wq : ((m == 1) ? Wk : Wv);
            wt[(size_t)m * NW + (size_t)n * ADIM + k] = (_Float16)W[k * HD + n];
        }
    }
}

// ---------------------------------------------------------------------------
// Kernel 2: fused QKV projection. 256 threads = 8 waves; each wave owns a
// 16-token tile (block covers 128 tokens). The W K-chunk (3*96 rows x 32 K,
// 18 KB f16) is shared by all waves: staged in LDS by a double-buffered
// TDM pipeline (copy of chunk i+1 overlaps the 18 WMMAs of chunk i).
// ---------------------------------------------------------------------------
__global__ __launch_bounds__(256)
void attn_qkv_kernel(const _Float16* __restrict__ xh,
                     const _Float16* __restrict__ wt,
                     const float* __restrict__ bq,
                     const float* __restrict__ bk,
                     const float* __restrict__ bv,
                     _Float16* __restrict__ qh,
                     _Float16* __restrict__ kh,
                     _Float16* __restrict__ vT) {
    __shared__ _Float16 wbuf[2][3 * HD * 32];   // 2 x 18 KB

    const int lane = threadIdx.x & 31;
    const int wave = threadIdx.x >> 5;
    const int m0   = blockIdx.x * 128 + wave * 16;   // global token tile base

    v8f acc[3][6];
#pragma unroll
    for (int m = 0; m < 3; ++m)
#pragma unroll
        for (int t = 0; t < 6; ++t) acc[m][t] = v8f{};

    const int NCHUNK = ADIM / 32;   // 24

#if ATTN_HAS_TDM
    if (wave == 0)
        tdm_load_2d_f16(lds_off(wbuf[0]), wt, 32, 3 * HD, ADIM);
#endif

    for (int kc = 0; kc < NCHUNK; ++kc) {
        const int k0  = kc * 32;
        const int cur = kc & 1;
#if ATTN_HAS_TDM
        if (wave == 0) __builtin_amdgcn_s_wait_tensorcnt(0);
        __syncthreads();            // wbuf[cur] ready for everyone
        if (wave == 0 && kc + 1 < NCHUNK)
            tdm_load_2d_f16(lds_off(wbuf[cur ^ 1]), wt + (size_t)(k0 + 32),
                            32, 3 * HD, ADIM);
#else
        __syncthreads();
        coop_copy_rows(wbuf[cur], wt + (size_t)k0, 3 * HD, 32, ADIM,
                       threadIdx.x, blockDim.x);
        __syncthreads();
#endif
        v16h a = load_a(xh, ADIM, m0, k0, lane);
#pragma unroll
        for (int m = 0; m < 3; ++m) {
#pragma unroll
            for (int t = 0; t < 6; ++t) {
                v16h b = load_b(wbuf[cur], 32, m * HD + t * 16, 0, lane);
                acc[m][t] = wmma16(a, b, acc[m][t]);
            }
        }
    }

    // C layout: lane holds column n = t*16 + (lane&15), rows mb..mb+7.
    const int nl = lane & 15;
    const int mb = m0 + ((lane >> 4) << 3);
    const int b  = m0 / ASEQ;
    const int s  = mb - b * ASEQ;
    _Float16* vTb = vT + (size_t)b * HD * ASEQ;
#pragma unroll
    for (int t = 0; t < 6; ++t) {
        const int n = t * 16 + nl;
        const float bqv = bq[n], bkv = bk[n], bvv = bv[n];
        h8 vv;
#pragma unroll
        for (int i = 0; i < 8; ++i) {
            qh[(size_t)(mb + i) * HD + n] = (_Float16)(acc[0][t][i] + bqv);
            kh[(size_t)(mb + i) * HD + n] = (_Float16)(acc[1][t][i] + bkv);
            vv[i] = (_Float16)(acc[2][t][i] + bvv);
        }
        *(h8*)(vTb + (size_t)n * ASEQ + s) = vv;   // contiguous 16B store
    }
}

// ---------------------------------------------------------------------------
// Kernel 3: per-KEY-column softmax statistics (softmax is over the q axis!).
// Each wave owns 16 key columns; streams all 256 q-tiles through 3 WMMAs,
// maintaining an online (max, sum-exp) per column directly in the C layout.
// ---------------------------------------------------------------------------
__global__ __launch_bounds__(128)
void attn_colstats_kernel(const _Float16* __restrict__ qh,
                          const _Float16* __restrict__ kh,
                          float* __restrict__ colmax,
                          float* __restrict__ colsum) {
    const int lane = threadIdx.x & 31;
    const int wave = threadIdx.x >> 5;
    const int kt   = blockIdx.x * 4 + wave;
    const int b    = kt / (ASEQ / 16);
    const int kc0  = (kt - b * (ASEQ / 16)) * 16;

    const _Float16* qb  = qh + (size_t)b * ASEQ * HD;
    const _Float16* kbm = kh + (size_t)b * ASEQ * HD;

    // K tile is fixed for this wave: 3 B fragments over d = 0,32,64.
    v16h bk0 = load_b(kbm, HD, kc0, 0,  lane);
    v16h bk1 = load_b(kbm, HD, kc0, 32, lane);
    v16h bk2 = load_b(kbm, HD, kc0, 64, lane);

    float mrun = -1e30f, zrun = 0.0f;
    for (int q0 = 0; q0 < ASEQ; q0 += 16) {
        __builtin_prefetch(qb + (size_t)(q0 + 16) * HD, 0, 1);
        v16h a0 = load_a(qb, HD, q0, 0,  lane);
        v16h a1 = load_a(qb, HD, q0, 32, lane);
        v16h a2 = load_a(qb, HD, q0, 64, lane);
        v8f c = v8f{};
        c = wmma16(a0, bk0, c);
        c = wmma16(a1, bk1, c);
        c = wmma16(a2, bk2, c);
        float tm = mrun;
#pragma unroll
        for (int i = 0; i < 8; ++i) tm = fmaxf(tm, c[i] * QKSCALE);
        float zs = 0.0f;
#pragma unroll
        for (int i = 0; i < 8; ++i) zs += __expf(c[i] * QKSCALE - tm);
        zrun = zrun * __expf(mrun - tm) + zs;
        mrun = tm;
    }
    // Merge the two M-halves (lane L holds rows i, lane L+16 holds rows i+8).
    const float mo = __shfl_xor(mrun, 16);
    const float zo = __shfl_xor(zrun, 16);
    const float mm = fmaxf(mrun, mo);
    const float zz = zrun * __expf(mrun - mm) + zo * __expf(mo - mm);
    if (lane < 16) {
        colmax[b * ASEQ + kc0 + lane] = mm;
        colsum[b * ASEQ + kc0 + lane] = zz;
    }
}

// ---------------------------------------------------------------------------
// Kernel 4: out = P @ v with P recomputed on the fly. 256 threads = 8 waves;
// each wave owns a 16-query tile. Every wave iterates the SAME key chunks,
// so the K tile (32x96) and V^T tile (96x32) are TDM-staged into LDS with a
// double buffer. Per chunk:
//   - recompute two 16x16 score tiles (3 WMMAs each) from LDS K
//   - P = exp(s - m_col)/Z_col, staged as f16 16x32 tile in wave-private LDS
//   - reload as A fragment, 6 WMMAs against LDS V^T (K = key tokens, N = d)
// ---------------------------------------------------------------------------
__global__ __launch_bounds__(256)
void attn_out_kernel(const _Float16* __restrict__ qh,
                     const _Float16* __restrict__ kh,
                     const _Float16* __restrict__ vT,
                     const float* __restrict__ colmax,
                     const float* __restrict__ colsum,
                     float* __restrict__ out) {
    __shared__ _Float16 pbuf[8][16 * 32];   //  8 KB, wave-private staging
    __shared__ _Float16 kbuf[2][32 * HD];   // 12 KB, K tile double buffer
    __shared__ _Float16 vbuf[2][HD * 32];   // 12 KB, V^T tile double buffer

    const int lane = threadIdx.x & 31;
    const int wave = threadIdx.x >> 5;
    const int qt   = blockIdx.x * 8 + wave;
    const int b    = qt / (ASEQ / 16);
    const int q0   = (qt - b * (ASEQ / 16)) * 16;

    const _Float16* qb  = qh + (size_t)b * ASEQ * HD;
    const _Float16* kbm = kh + (size_t)b * ASEQ * HD;
    const _Float16* vb  = vT + (size_t)b * HD * ASEQ;
    const float*    cm  = colmax + b * ASEQ;
    const float*    cs  = colsum + b * ASEQ;

    // Q tile fixed for this wave.
    v16h aq0 = load_a(qb, HD, q0, 0,  lane);
    v16h aq1 = load_a(qb, HD, q0, 32, lane);
    v16h aq2 = load_a(qb, HD, q0, 64, lane);

    v8f acc[6];
#pragma unroll
    for (int t = 0; t < 6; ++t) acc[t] = v8f{};

    const int nl = lane & 15;
    const int hb = lane >> 4;
    _Float16* myp = pbuf[wave];
    const int NCHUNK = ASEQ / 32;   // 128

#if ATTN_HAS_TDM
    if (wave == 0) {
        // K chunk rows are contiguous in memory: one 6KB 1-D transfer.
        tdm_load_2d_f16(lds_off(kbuf[0]), kbm, 32 * HD, 1, 32 * HD);
        tdm_load_2d_f16(lds_off(vbuf[0]), vb, 32, HD, ASEQ);
    }
#endif

    for (int kc = 0; kc < NCHUNK; ++kc) {
        const int k0  = kc * 32;
        const int cur = kc & 1;
#if ATTN_HAS_TDM
        if (wave == 0) __builtin_amdgcn_s_wait_tensorcnt(0);
        __syncthreads();            // staged tiles visible to all waves
        if (wave == 0 && kc + 1 < NCHUNK) {
            tdm_load_2d_f16(lds_off(kbuf[cur ^ 1]),
                            kbm + (size_t)(k0 + 32) * HD, 32 * HD, 1, 32 * HD);
            tdm_load_2d_f16(lds_off(vbuf[cur ^ 1]),
                            vb + (size_t)(k0 + 32), 32, HD, ASEQ);
        }
#else
        __syncthreads();
        coop_copy_rows(kbuf[cur], kbm + (size_t)k0 * HD, 32, HD, HD,
                       threadIdx.x, blockDim.x);
        coop_copy_rows(vbuf[cur], vb + (size_t)k0, HD, 32, ASEQ,
                       threadIdx.x, blockDim.x);
        __syncthreads();
#endif
#pragma unroll
        for (int half = 0; half < 2; ++half) {
            const int kloc = half * 16;           // local key index in chunk
            v16h b0 = load_b(kbuf[cur], HD, kloc, 0,  lane);
            v16h b1 = load_b(kbuf[cur], HD, kloc, 32, lane);
            v16h b2 = load_b(kbuf[cur], HD, kloc, 64, lane);
            v8f sc = v8f{};
            sc = wmma16(aq0, b0, sc);
            sc = wmma16(aq1, b1, sc);
            sc = wmma16(aq2, b2, sc);
            const int n  = k0 + kloc + nl;
            const float mc = cm[n];
            const float rz = 1.0f / cs[n];
            const int mloc = hb * 8;
#pragma unroll
            for (int i = 0; i < 8; ++i) {
                const float p = __expf(sc[i] * QKSCALE - mc) * rz;
                myp[(mloc + i) * 32 + kloc + nl] = (_Float16)p;
            }
        }
        // Wave-private LDS: DS ops from the same wave stay in order.
        v16h ap = load_a(myp, 32, 0, 0, lane);
#pragma unroll
        for (int t = 0; t < 6; ++t) {
            v16h bvf = load_b(vbuf[cur], 32, t * 16, 0, lane);  // n=d, K=token
            acc[t] = wmma16(ap, bvf, acc[t]);
        }
    }

    // Store out f32 [b][q][96]; lane holds column n, rows mb..mb+7.
    const int mb = q0 + hb * 8;
    float* ob = out + (size_t)b * ASEQ * HD;
#pragma unroll
    for (int t = 0; t < 6; ++t) {
        const int n = t * 16 + nl;
#pragma unroll
        for (int i = 0; i < 8; ++i)
            ob[(size_t)(mb + i) * HD + n] = acc[t][i];
    }
}

// ---------------------------------------------------------------------------
extern "C" void kernel_launch(void* const* d_in, const int* in_sizes, int n_in,
                              void* d_out, int out_size, void* d_ws, size_t ws_size,
                              hipStream_t stream) {
    const float* x  = (const float*)d_in[0];
    const float* Wq = (const float*)d_in[1];
    const float* bq = (const float*)d_in[2];
    const float* Wk = (const float*)d_in[3];
    const float* bk = (const float*)d_in[4];
    const float* Wv = (const float*)d_in[5];
    const float* bv = (const float*)d_in[6];
    float* out = (float*)d_out;

    // Workspace carve-up (all chunks 256B-aligned by construction). ~35.2 MB.
    char* ws = (char*)d_ws;
    size_t off = 0;
    _Float16* xh = (_Float16*)(ws + off); off += (size_t)NTOK * ADIM * 2;       // 25.2 MB
    _Float16* wt = (_Float16*)(ws + off); off += (size_t)3 * ADIM * HD * 2;     // 442 KB
    _Float16* qh = (_Float16*)(ws + off); off += (size_t)NTOK * HD * 2;         // 3.1 MB
    _Float16* kh = (_Float16*)(ws + off); off += (size_t)NTOK * HD * 2;         // 3.1 MB
    _Float16* vT = (_Float16*)(ws + off); off += (size_t)NTOK * HD * 2;         // 3.1 MB
    float* colmax = (float*)(ws + off);   off += (size_t)NTOK * 4;              // 64 KB
    float* colsum = (float*)(ws + off);   off += (size_t)NTOK * 4;              // 64 KB

    attn_convert_kernel<<<2048, 256, 0, stream>>>(x, Wq, Wk, Wv, xh, wt);
    attn_qkv_kernel<<<NTOK / 128, 256, 0, stream>>>(xh, wt, bq, bk, bv, qh, kh, vT);
    attn_colstats_kernel<<<NTOK / 64, 128, 0, stream>>>(qh, kh, colmax, colsum);
    attn_out_kernel<<<NTOK / 128, 256, 0, stream>>>(qh, kh, vT, colmax, colsum, out);
}
